// WindowAttention_12421045420428
// MI455X (gfx1250) — matrix operand order repaired and checked by reading, not compiled, not run on hardware
//
#include <hip/hip_runtime.h>
#include <hip/hip_bf16.h>
#include <math.h>

// ---------------------------------------------------------------------------
// Fused Swin window attention for gfx1250 (MI455X), wave32 + WMMA f16.
//   Bw=16384 windows, N=49 (padded to 64), C=128, H=4 heads, hd=32, nW=1024.
// One workgroup (256 threads = 8 waves) processes one full window:
//   x -> qkv -> softmax(qk^T + bias + mask) -> @v -> proj   entirely in LDS.
// ---------------------------------------------------------------------------

typedef __attribute__((ext_vector_type(16))) _Float16 v16h;
typedef __attribute__((ext_vector_type(8)))  _Float16 v8h;
typedef __attribute__((ext_vector_type(8)))  float    v8f;
typedef __attribute__((ext_vector_type(4)))  float    v4f;

#define NW       49
#define NP       64
#define CDIM     128
#define HEADS    4
#define HD       32
#define NWIN     1024
#define SCALE    0.17677669529663687f  /* 1/sqrt(32) */

// ---- LDS layout (bytes) ---------------------------------------------------
#define OFF_XS   0            /* x window f16 [64][128]; reused as attn-out  */
#define OFF_QS   16384        /* q f16 [64][128] (pre-scaled)                */
#define OFF_KS   32768        /* k f16 [64][128]                             */
#define OFF_VT   49152        /* v^T f16 [128][64]                           */
#define OFF_SS   65536        /* softmax scratch f32 [8 waves][32][64]       */
#define OFF_PS   131072       /* P (exp, unnormalized) f16 [8][32][64]       */
#define OFF_RS   163840       /* 1/rowsum f32 [8][32]                        */
#define LDS_BYTES 164864

// ---- workspace layout (bytes) ---------------------------------------------
#define WS_WQ    0            /* qkv_w as f16, 384*128                       */
#define WS_WP    98304        /* proj_w as f16, 128*128                      */
#define WS_BP    131072       /* padded bias f32 [4][64][64], -1e30 pad      */
// total = 196608 bytes

static __device__ __forceinline__ v8f wmma_f16(v16h a, v16h b, v8f c) {
  return __builtin_amdgcn_wmma_f32_16x16x32_f16(false, a, false, b,
                                                (short)0, c, false, false);
}

// Load a 16x32 f16 A-fragment from a row-major [rows][ld] buffer.
// lane L: m = base_m + (L&15); halves e: K = 16*(L>>4) + e.
static __device__ __forceinline__ v16h load_a_frag(const _Float16* p, int row,
                                                   int ld, int kbase,
                                                   int ln15, int lh) {
  const _Float16* r = p + (row + ln15) * ld + kbase + lh * 16;
  return *(const v16h*)(r);
}

// Load a 32x16 f16 B-fragment where memory is K-contiguous per output column:
// element (k,n) at p[(n)*ld + k].  lane L: n = base_n + (L&15),
// halves e: K = 16*(L>>4) + e  -> one 32B contiguous read.
static __device__ __forceinline__ v16h load_b_frag(const _Float16* p, int col,
                                                   int ld, int kbase,
                                                   int ln15, int lh) {
  return *(const v16h*)(p + (col + ln15) * ld + kbase + lh * 16);
}

// ---------------------------------------------------------------------------
// Prep: f32->f16 weights, and gather padded relative-position bias.
// ---------------------------------------------------------------------------
__global__ void swin_prep(const float* __restrict__ qkv_w,
                          const float* __restrict__ proj_w,
                          const float* __restrict__ rpb,
                          const int*   __restrict__ rel,
                          _Float16* __restrict__ wq,
                          _Float16* __restrict__ wp,
                          float* __restrict__ biaspad) {
  int i = blockIdx.x * blockDim.x + threadIdx.x;
  if (i < 3 * CDIM * CDIM) wq[i] = (_Float16)qkv_w[i];
  if (i < CDIM * CDIM)     wp[i] = (_Float16)proj_w[i];
  if (i < HEADS * NP * NP) {
    int h = i >> 12, rest = i & 4095, r = rest >> 6, c = rest & 63;
    float v = -1.0e30f;
    if (r < NW && c < NW) v = rpb[rel[r * NW + c] * HEADS + h];
    biaspad[i] = v;
  }
}

// ---------------------------------------------------------------------------
// Main fused kernel: one block per window.
// ---------------------------------------------------------------------------
__global__ __launch_bounds__(256) void swin_attn(
    const float* __restrict__ x, const float* __restrict__ mask,
    const float* __restrict__ qkv_b, const float* __restrict__ proj_b,
    const _Float16* __restrict__ wq, const _Float16* __restrict__ wp,
    const float* __restrict__ biaspad, float* __restrict__ out) {

  __shared__ __align__(16) unsigned char smem[LDS_BYTES];
  _Float16* xs  = (_Float16*)(smem + OFF_XS);
  _Float16* qs  = (_Float16*)(smem + OFF_QS);
  _Float16* ks  = (_Float16*)(smem + OFF_KS);
  _Float16* vts = (_Float16*)(smem + OFF_VT);

  const int tid  = threadIdx.x;
  const int win  = blockIdx.x;
  const int wv   = tid >> 5;
  const int lane = tid & 31;
  const int ln15 = lane & 15;
  const int lh   = lane >> 4;

  // ---- Phase A: stage x window into LDS as f16, zero-pad rows 49..63 ------
  for (int i = tid; i < (NP * CDIM) / 2; i += 256) {
    int m = i >> 6, c = (i & 63) * 2;
    float a0 = 0.f, a1 = 0.f;
    if (m < NW) {
      const float* px = x + ((size_t)win * NW + m) * CDIM + c;
      a0 = px[0]; a1 = px[1];
    }
    xs[m * CDIM + c]     = (_Float16)a0;
    xs[m * CDIM + c + 1] = (_Float16)a1;
  }
  __syncthreads();

  // ---- Phase B: qkv = x @ qkv_w^T + b.  96 tiles; wave owns 3 N-tiles -----
  for (int tt = 0; tt < 3; ++tt) {
    int tn = wv * 3 + tt;
    int n0 = tn * 16 + ln15;                       // output column in [0,384)
    v16h bf[4];
#pragma unroll
    for (int kk = 0; kk < 4; ++kk)
      bf[kk] = load_b_frag(wq, tn * 16, CDIM, kk * 32, ln15, lh);
    float bias = qkv_b[n0];
#pragma unroll
    for (int tm = 0; tm < 4; ++tm) {
      v8f acc = {};
#pragma unroll
      for (int kk = 0; kk < 4; ++kk) {
        v16h af = load_a_frag(xs, tm * 16, CDIM, kk * 32, ln15, lh);
        acc = wmma_f16(af, bf[kk], acc);
      }
      if (n0 < CDIM) {                              // q (pre-scaled)
#pragma unroll
        for (int r = 0; r < 8; ++r) {
          int m = tm * 16 + r + lh * 8;
          qs[m * CDIM + n0] = (_Float16)((acc[r] + bias) * SCALE);
        }
      } else if (n0 < 2 * CDIM) {                   // k
        int c = n0 - CDIM;
#pragma unroll
        for (int r = 0; r < 8; ++r) {
          int m = tm * 16 + r + lh * 8;
          ks[m * CDIM + c] = (_Float16)(acc[r] + bias);
        }
      } else {                                      // v, stored transposed
        int c = n0 - 2 * CDIM;
#pragma unroll
        for (int r = 0; r < 8; ++r) {
          int m = tm * 16 + r + lh * 8;
          vts[c * NP + m] = (_Float16)(acc[r] + bias);
        }
      }
    }
  }
  __syncthreads();

  // ---- Phase C: S = q k^T + bias + mask.  2 waves per head ----------------
  const int h  = wv >> 1;          // head
  const int mh = wv & 1;           // which 32-row half of the window
  float* ssc = (float*)(smem + OFF_SS) + wv * 2048;          // [32][64]
  const float* bp = biaspad + h * NP * NP;
  const float* mk = mask + (size_t)(win & (NWIN - 1)) * NW * NW;
#pragma unroll
  for (int tm2 = 0; tm2 < 2; ++tm2) {
    int mt = mh * 2 + tm2;
    v16h af = load_a_frag(qs, mt * 16, CDIM, h * HD, ln15, lh);   // K = hd = 32
#pragma unroll
    for (int tn = 0; tn < 4; ++tn) {
      int n = tn * 16 + ln15;
      v16h bfr = load_b_frag(ks, tn * 16, CDIM, h * HD, ln15, lh);
      v8f acc = {};
      acc = wmma_f16(af, bfr, acc);
#pragma unroll
      for (int r = 0; r < 8; ++r) {
        int m = mt * 16 + r + lh * 8;
        float a = acc[r] + bp[m * NP + n];
        if (m < NW && n < NW) a += mk[m * NW + n];
        ssc[(tm2 * 16 + r + lh * 8) * NP + n] = a;
      }
    }
  }
  __syncthreads();

  // ---- Phase D: row softmax (one lane per row); P = exp (unnormalized) ----
  {
    float* srow = ssc + lane * NP;
    float mx = -3.0e38f;
#pragma unroll
    for (int j = 0; j < NP; j += 4) {
      v4f t = *(const v4f*)(srow + j);
      mx = fmaxf(mx, fmaxf(fmaxf(t[0], t[1]), fmaxf(t[2], t[3])));
    }
    _Float16* prow = (_Float16*)(smem + OFF_PS) + wv * 2048 + lane * NP;
    float sum = 0.f;
#pragma unroll
    for (int j = 0; j < NP; j += 4) {
      v4f t = *(const v4f*)(srow + j);
#pragma unroll
      for (int u = 0; u < 4; ++u) {
        float e = __expf(t[u] - mx);
        sum += e;
        prow[j + u] = (_Float16)e;
      }
    }
    ((float*)(smem + OFF_RS))[wv * 32 + lane] = 1.0f / sum;
  }
  __syncthreads();

  // ---- Phase E: O = P @ V (V^T in LDS -> K-contiguous B frags) ------------
  {
    const _Float16* ps = (const _Float16*)(smem + OFF_PS) + wv * 2048;
    const float* rs = (const float*)(smem + OFF_RS) + wv * 32;
    _Float16* os = (_Float16*)(smem + OFF_XS);   // reuse x buffer for output
#pragma unroll
    for (int tm2 = 0; tm2 < 2; ++tm2) {
#pragma unroll
      for (int td = 0; td < 2; ++td) {
        v8f acc = {};
#pragma unroll
        for (int kk = 0; kk < 2; ++kk) {
          v16h af  = load_a_frag(ps, tm2 * 16, NP, kk * 32, ln15, lh);
          v16h bfr = load_b_frag(vts, h * HD + td * 16, NP, kk * 32, ln15, lh);
          acc = wmma_f16(af, bfr, acc);
        }
#pragma unroll
        for (int r = 0; r < 8; ++r) {
          int lr = tm2 * 16 + r + lh * 8;
          int m  = mh * 32 + lr;
          int c  = h * HD + td * 16 + ln15;
          os[m * CDIM + c] = (_Float16)(acc[r] * rs[lr]);
        }
      }
    }
  }
  __syncthreads();

  // ---- Phase F: out = O @ proj_w^T + proj_b; store 49 valid rows ----------
  {
    const _Float16* os = (const _Float16*)(smem + OFF_XS);
    int n0 = wv * 16 + ln15;                       // each wave owns one N-tile
    v16h bf[4];
#pragma unroll
    for (int kk = 0; kk < 4; ++kk)
      bf[kk] = load_b_frag(wp, wv * 16, CDIM, kk * 32, ln15, lh);
    float pb = proj_b[n0];
#pragma unroll
    for (int tm = 0; tm < 4; ++tm) {
      v8f acc = {};
#pragma unroll
      for (int kk = 0; kk < 4; ++kk) {
        v16h af = load_a_frag(os, tm * 16, CDIM, kk * 32, ln15, lh);
        acc = wmma_f16(af, bf[kk], acc);
      }
#pragma unroll
      for (int r = 0; r < 8; ++r) {
        int m = tm * 16 + r + lh * 8;
        if (m < NW)
          out[((size_t)win * NW + m) * CDIM + n0] = acc[r] + pb;
      }
    }
  }
}

// ---------------------------------------------------------------------------
extern "C" void kernel_launch(void* const* d_in, const int* in_sizes, int n_in,
                              void* d_out, int out_size, void* d_ws, size_t ws_size,
                              hipStream_t stream) {
  const float* x      = (const float*)d_in[0];
  const float* mask   = (const float*)d_in[1];
  const float* qkv_w  = (const float*)d_in[2];
  const float* qkv_b  = (const float*)d_in[3];
  const float* proj_w = (const float*)d_in[4];
  const float* proj_b = (const float*)d_in[5];
  const float* rpb    = (const float*)d_in[6];
  const int*   rel    = (const int*)d_in[7];
  float* out = (float*)d_out;

  _Float16* wq = (_Float16*)((char*)d_ws + WS_WQ);
  _Float16* wp = (_Float16*)((char*)d_ws + WS_WP);
  float*    bp = (float*)((char*)d_ws + WS_BP);

  swin_prep<<<192, 256, 0, stream>>>(qkv_w, proj_w, rpb, rel, wq, wp, bp);

  const int Bw = in_sizes[0] / (NW * CDIM);       // 16384
  swin_attn<<<Bw, 256, 0, stream>>>(x, mask, qkv_b, proj_b, wq, wp, bp, out);
}
